// Attention_28776280883773
// MI455X (gfx1250) — matrix operand order repaired and checked by reading, not compiled, not run on hardware
//
#include <hip/hip_runtime.h>
#include <hip/hip_bf16.h>

typedef __attribute__((ext_vector_type(16))) __bf16   v16bf;
typedef __attribute__((ext_vector_type(8)))  float    v8f;
typedef __attribute__((ext_vector_type(4)))  uint32_t u32x4;
typedef __attribute__((ext_vector_type(8)))  uint32_t u32x8;

#define DIM      512
#define HEADS    8
#define HEAD_DIM 64
#define NQ       4096
#define NKV      1024
#define BS       4
#define ATT_SCALE 0.125f   // 64^-0.5, folded into q projection

// ---------------------------------------------------------------------------
// WMMA helpers (CDNA5 wave32, v_wmma_f32_16x16x32_bf16)
// ---------------------------------------------------------------------------
__device__ __forceinline__ v8f zero8() {
  v8f z;
#pragma unroll
  for (int i = 0; i < 8; ++i) z[i] = 0.0f;
  return z;
}

__device__ __forceinline__ v8f wmma_bf16(v16bf a, v16bf b, v8f c) {
  // (neg_a, A, neg_b, B, c_mod, C, reuse_a, reuse_b)
  return __builtin_amdgcn_wmma_f32_16x16x32_bf16(false, a, false, b, (short)0,
                                                 c, false, false);
}

// A-matrix 16x32 bf16 fragment. Lane L: m = L&15, h = L>>4.
// elems 0..7  <- A[row0+m][k0 + h*8     .. +7]
// elems 8..15 <- A[row0+m][k0 + 16+h*8  .. +7]
__device__ __forceinline__ v16bf load_a16x32(const __bf16* base, int lda,
                                             int row0, int k0) {
  const int lane = threadIdx.x & 31;
  const int m = lane & 15, h = lane >> 4;
  const __bf16* p = base + (size_t)(row0 + m) * lda + k0 + h * 8;
  union { uint4 u[2]; v16bf v; } t;
  t.u[0] = *reinterpret_cast<const uint4*>(p);
  t.u[1] = *reinterpret_cast<const uint4*>(p + 16);
  return t.v;
}

// B-matrix 32x16 bf16 fragment from TRANSPOSED storage T[n][k] (ldt elems).
// Lane L: n = L&15, h = L>>4; elems 0..7 <- T[n0+n][k0 + h*8 .. +7],
// elems 8..15 <- T[n0+n][k0 + 16+h*8 .. +7]
__device__ __forceinline__ v16bf load_b32x16_t(const __bf16* baseT, int ldt,
                                               int n0, int k0) {
  const int lane = threadIdx.x & 31;
  const int n = lane & 15, h = lane >> 4;
  const __bf16* p = baseT + (size_t)(n0 + n) * ldt + k0 + h * 8;
  union { uint4 u[2]; v16bf v; } t;
  t.u[0] = *reinterpret_cast<const uint4*>(p);
  t.u[1] = *reinterpret_cast<const uint4*>(p + 16);
  return t.v;
}

// B-matrix 32x16 fragment from row-major B[k][n] (scalar gather, LDS use).
__device__ __forceinline__ v16bf load_b32x16_rm(const __bf16* base, int ldb,
                                                int k0, int n0) {
  const int lane = threadIdx.x & 31;
  const int n = lane & 15, h = lane >> 4;
  v16bf v;
#pragma unroll
  for (int e = 0; e < 16; ++e)
    v[e] = base[(size_t)(k0 + h * 8 + e + (e >= 8 ? 8 : 0)) * ldb + n0 + n];
  return v;
}

// ---------------------------------------------------------------------------
// Generic bf16 GEMM: C(MxN) = (A(MxK) @ B(KxN)) * oscale [+ bias], f32 acc.
// Block = 256 threads (8 waves), tile 128x64, K-step 32, B staged transposed
// in LDS so B-fragments are two 16B ds_loads per lane.
// ---------------------------------------------------------------------------
template <typename OutT>
__global__ __launch_bounds__(256) void gemm_bf16(const __bf16* __restrict__ A,
                                                 const __bf16* __restrict__ B,
                                                 OutT* __restrict__ C,
                                                 const float* __restrict__ bias,
                                                 float oscale,
                                                 int M, int N, int K) {
  __shared__ __bf16 Bt[64 * 32];  // [n][k], ld = 32 (64B rows, 16B aligned)
  const int tid  = threadIdx.x;
  const int lane = tid & 31;
  const int wave = tid >> 5;
  const int row0 = blockIdx.y * 128 + wave * 16;
  const int col0 = blockIdx.x * 64;

  v8f acc[4];
#pragma unroll
  for (int i = 0; i < 4; ++i) acc[i] = zero8();

  const int kk = tid >> 3;        // 0..31 : k within tile
  const int n8 = (tid & 7) * 8;   // 0..56 : n group of 8

  for (int kb = 0; kb < K; kb += 32) {
    // Stage B[kb..kb+32) x [col0..col0+64) transposed into LDS.
    {
      const __bf16* src = B + (size_t)(kb + kk) * N + col0 + n8;
      uint4 d = *reinterpret_cast<const uint4*>(src);  // 8 bf16, coalesced
      const __bf16* ds = reinterpret_cast<const __bf16*>(&d);
#pragma unroll
      for (int i = 0; i < 8; ++i) Bt[(n8 + i) * 32 + kk] = ds[i];
      if (kb + 32 < K)
        __builtin_prefetch(B + (size_t)(kb + 32 + kk) * N + col0 + n8, 0, 1);
    }
    __syncthreads();

    v16bf a = load_a16x32(A, K, row0, kb);
#pragma unroll
    for (int nt = 0; nt < 4; ++nt) {
      v16bf b = load_b32x16_t(Bt, 32, nt * 16, 0);
      acc[nt] = wmma_bf16(a, b, acc[nt]);
    }
    __syncthreads();
  }

  // C layout: row = row0 + (lane>>4)*8 + r, col = col0 + nt*16 + (lane&15)
  const int nn = lane & 15, h = lane >> 4;
#pragma unroll
  for (int nt = 0; nt < 4; ++nt) {
    const int col = col0 + nt * 16 + nn;
    const float bv = bias ? bias[col] : 0.0f;
#pragma unroll
    for (int r = 0; r < 8; ++r) {
      const int row = row0 + h * 8 + r;
      C[(size_t)row * N + col] = (OutT)(acc[nt][r] * oscale + bv);
    }
  }
}

// ---------------------------------------------------------------------------
// f32 -> bf16 convert (count multiple of 4)
// ---------------------------------------------------------------------------
__global__ void f32_to_bf16(const float* __restrict__ in,
                            __bf16* __restrict__ out, int count) {
  const int i = (blockIdx.x * blockDim.x + threadIdx.x) * 4;
  if (i + 3 < count) {
    float4 f = *reinterpret_cast<const float4*>(in + i);
    out[i]     = (__bf16)f.x;
    out[i + 1] = (__bf16)f.y;
    out[i + 2] = (__bf16)f.z;
    out[i + 3] = (__bf16)f.w;
  }
}

// sr_w (O=512, I=512, 2, 2) -> srw[(dy*2+dx)*512*512 + i*512 + o]  (bf16)
__global__ void permute_srw(const float* __restrict__ w,
                            __bf16* __restrict__ out) {
  const int idx = blockIdx.x * blockDim.x + threadIdx.x;  // 512*512*4
  const int o   = idx >> 11;          // / (512*4)
  const int rem = idx & 2047;
  const int i   = rem >> 2;
  const int dd  = rem & 3;
  out[(size_t)dd * (512 * 512) + (size_t)i * 512 + o] = (__bf16)w[idx];
}

// Build conv-as-GEMM patch matrix: patch[(b*1024+p)][dd*512+i] =
//   x_bf[b*4096 + (2py+dy)*64 + 2px+dx][i]
__global__ void gather_patches(const __bf16* __restrict__ xbf,
                               __bf16* __restrict__ patch) {
  const int prow = blockIdx.x;   // 0..4095
  const int dd   = blockIdx.y;   // 0..3
  const int b  = prow >> 10;
  const int p  = prow & 1023;
  const int py = p >> 5, px = p & 31;
  const int dy = dd >> 1, dx = dd & 1;
  const int srow = b * 4096 + (2 * py + dy) * 64 + (2 * px + dx);
  const __bf16* src = xbf + (size_t)srow * 512;
  __bf16* dst = patch + (size_t)prow * 2048 + dd * 512;
  const int t = threadIdx.x;  // 128 threads x 4 bf16 (8B)
  *reinterpret_cast<uint2*>(dst + t * 4) =
      *reinterpret_cast<const uint2*>(src + t * 4);
}

// LayerNorm over 512 channels, fusing conv bias: out = LN(in + sr_b)*w + b
__global__ __launch_bounds__(256) void layernorm_bias(
    const float* __restrict__ in, const float* __restrict__ sr_b,
    const float* __restrict__ ln_w, const float* __restrict__ ln_b,
    __bf16* __restrict__ out) {
  __shared__ float red[8], red2[8];
  const int row = blockIdx.x;
  const int t = threadIdx.x;
  const float* x = in + (size_t)row * 512;
  float v0 = x[t] + sr_b[t];
  float v1 = x[t + 256] + sr_b[t + 256];
  float s = v0 + v1, s2 = v0 * v0 + v1 * v1;
#pragma unroll
  for (int d = 16; d >= 1; d >>= 1) {
    s  += __shfl_xor(s, d);
    s2 += __shfl_xor(s2, d);
  }
  if ((t & 31) == 0) { red[t >> 5] = s; red2[t >> 5] = s2; }
  __syncthreads();
  float tot = 0.f, tot2 = 0.f;
#pragma unroll
  for (int i = 0; i < 8; ++i) { tot += red[i]; tot2 += red2[i]; }
  const float mean = tot * (1.0f / 512.0f);
  const float var  = tot2 * (1.0f / 512.0f) - mean * mean;
  const float rstd = rsqrtf(var + 1e-5f);
  out[(size_t)row * 512 + t] =
      (__bf16)((v0 - mean) * rstd * ln_w[t] + ln_b[t]);
  out[(size_t)row * 512 + t + 256] =
      (__bf16)((v1 - mean) * rstd * ln_w[t + 256] + ln_b[t + 256]);
}

// ---------------------------------------------------------------------------
// Flash attention: 1 wave per (b, h, 16-query tile). 32-key blocks.
// Q:(BS*NQ,512) bf16 pre-scaled by ATT_SCALE;
// KV:(BS*NKV,1024) bf16 (k at col h*64, v at 512+h*64).
// V block staged LDS-side by the Tensor Data Mover (TDM), overlapped with
// the S = QK^T WMMAs and softmax, fenced with s_wait_tensorcnt.
// ---------------------------------------------------------------------------
__global__ __launch_bounds__(32) void attention(const __bf16* __restrict__ Q,
                                                const __bf16* __restrict__ KV,
                                                __bf16* __restrict__ Onet) {
  __shared__ __bf16 Vs[32 * 64];  // TDM-staged V block [key][d]
  __shared__ __bf16 Ps[16 * 32];  // P in row-major for A-fragment reload
  const int lane = threadIdx.x;
  const int qt = blockIdx.x, h = blockIdx.y, b = blockIdx.z;

  const int qrow0 = b * NQ + qt * 16;
  const v16bf aq0 = load_a16x32(Q, DIM, qrow0, h * HEAD_DIM);
  const v16bf aq1 = load_a16x32(Q, DIM, qrow0, h * HEAD_DIM + 32);

  const __bf16* Kbase = KV + (size_t)b * NKV * (2 * DIM) + h * HEAD_DIM;
  const __bf16* Vbase = Kbase + DIM;

  // TDM descriptor group 1 (invariant): 2D tile, 2-byte elements,
  // tile 64 x 32, row stride = 1024 elements (KV row pitch).
  u32x8 g1;
  g1[0] = 1u << 16;           // data_size = 2 bytes
  g1[1] = 64u << 16;          // tensor_dim0[15:0] = 64
  g1[2] = 32u << 16;          // tensor_dim0[31:16]=0, tensor_dim1 = 32
  g1[3] = 64u << 16;          // tensor_dim1 hi = 0, tile_dim0 = 64
  g1[4] = 32u;                // tile_dim1 = 32, tile_dim2 = 0
  g1[5] = 1024u;              // tensor_dim0_stride = 1024 elements
  g1[6] = 0u;
  g1[7] = 0u;
  const uint32_t lds_off = (uint32_t)(uintptr_t)(void*)Vs;  // LDS aperture low bits

  v8f o[4];
#pragma unroll
  for (int i = 0; i < 4; ++i) o[i] = zero8();
  float mrow[8], lrow[8];
#pragma unroll
  for (int r = 0; r < 8; ++r) { mrow[r] = -3.0e38f; lrow[r] = 0.0f; }

  const int hh = lane >> 4, nn = lane & 15;

  for (int kb = 0; kb < NKV; kb += 32) {
    // Kick off TDM: V[kb..kb+32) x [0..64) -> Vs (async, TENSORcnt).
    {
      const uint64_t gaddr =
          (uint64_t)(uintptr_t)(Vbase + (size_t)kb * (2 * DIM));
      u32x4 g0;
      g0[0] = 1u;                                    // count = 1 descriptor
      g0[1] = lds_off;                               // lds_addr
      g0[2] = (uint32_t)gaddr;                       // global_addr[31:0]
      g0[3] = (uint32_t)(gaddr >> 32) | (2u << 30);  // addr[56:32], type=2
      asm volatile("tensor_load_to_lds %0, %1" ::"s"(g0), "s"(g1) : "memory");
    }

    // S = q @ k^T : row-major K is exactly transposed-B storage.
    v8f s0 = zero8(), s1 = zero8();
    s0 = wmma_bf16(aq0, load_b32x16_t(Kbase, 2 * DIM, kb,      0),  s0);
    s0 = wmma_bf16(aq1, load_b32x16_t(Kbase, 2 * DIM, kb,      32), s0);
    s1 = wmma_bf16(aq0, load_b32x16_t(Kbase, 2 * DIM, kb + 16, 0),  s1);
    s1 = wmma_bf16(aq1, load_b32x16_t(Kbase, 2 * DIM, kb + 16, 32), s1);

    // Online softmax (scale pre-folded into Q).
    // C layout: row = hh*8+r, col = nn; reduce within 16-lane halves.
    float p0[8], p1[8], corr[8];
#pragma unroll
    for (int r = 0; r < 8; ++r) {
      float mx = fmaxf(s0[r], s1[r]);
      mx = fmaxf(mx, __shfl_xor(mx, 1));
      mx = fmaxf(mx, __shfl_xor(mx, 2));
      mx = fmaxf(mx, __shfl_xor(mx, 4));
      mx = fmaxf(mx, __shfl_xor(mx, 8));
      const float mnew = fmaxf(mrow[r], mx);
      corr[r] = __expf(mrow[r] - mnew);
      mrow[r] = mnew;
      p0[r] = __expf(s0[r] - mnew);
      p1[r] = __expf(s1[r] - mnew);
      float ps = p0[r] + p1[r];
      ps += __shfl_xor(ps, 1);
      ps += __shfl_xor(ps, 2);
      ps += __shfl_xor(ps, 4);
      ps += __shfl_xor(ps, 8);
      lrow[r] = lrow[r] * corr[r] + ps;
    }
#pragma unroll
    for (int nt = 0; nt < 4; ++nt)
#pragma unroll
      for (int r = 0; r < 8; ++r) o[nt][r] *= corr[r];

    // C-layout -> A-layout via LDS (single wave: DS ops are in-order).
#pragma unroll
    for (int r = 0; r < 8; ++r) {
      const int row = hh * 8 + r;
      Ps[row * 32 + nn]      = (__bf16)p0[r];
      Ps[row * 32 + 16 + nn] = (__bf16)p1[r];
    }
    __syncthreads();  // single-wave WG: degenerates, forces dscnt ordering

    // Fence the TDM before reading Vs.
    asm volatile("s_wait_tensorcnt 0x0" ::: "memory");

    const v16bf ap = load_a16x32(Ps, 32, 0, 0);
#pragma unroll
    for (int nt = 0; nt < 4; ++nt) {
      const v16bf bv = load_b32x16_rm(Vs, 64, 0, nt * 16);
      o[nt] = wmma_bf16(ap, bv, o[nt]);
    }
    __syncthreads();
  }

  // Epilogue: normalize and store bf16 net (bs*n, 512).
#pragma unroll
  for (int nt = 0; nt < 4; ++nt) {
    const int col = h * HEAD_DIM + nt * 16 + nn;
#pragma unroll
    for (int r = 0; r < 8; ++r) {
      const int row = qrow0 + hh * 8 + r;
      Onet[(size_t)row * DIM + col] = (__bf16)(o[nt][r] / lrow[r]);
    }
  }
}

// ---------------------------------------------------------------------------
// Host orchestration
// ---------------------------------------------------------------------------
extern "C" void kernel_launch(void* const* d_in, const int* in_sizes, int n_in,
                              void* d_out, int out_size, void* d_ws,
                              size_t ws_size, hipStream_t stream) {
  (void)in_sizes; (void)n_in; (void)out_size; (void)ws_size;
  const float* x      = (const float*)d_in[0];
  const float* q_w    = (const float*)d_in[1];
  const float* kv_w   = (const float*)d_in[2];
  const float* proj_w = (const float*)d_in[3];
  const float* proj_b = (const float*)d_in[4];
  const float* sr_w   = (const float*)d_in[5];
  const float* sr_b   = (const float*)d_in[6];
  const float* ln_w   = (const float*)d_in[7];
  const float* ln_b   = (const float*)d_in[8];
  float* out = (float*)d_out;

  char* ws = (char*)d_ws;
  const size_t MiB = 1u << 20;
  __bf16* x_bf    = (__bf16*)(ws + 0 * MiB);    // 16 MiB (16384x512)
  __bf16* q_bf    = (__bf16*)(ws + 16 * MiB);   // 16 MiB (16384x512)
  __bf16* patch   = (__bf16*)(ws + 32 * MiB);   // 16 MiB (4096x2048) -> net
  float*  convout = (float*) (ws + 48 * MiB);   //  8 MiB (4096x512 f32)
  __bf16* kv_bf   = (__bf16*)(ws + 56 * MiB);   //  8 MiB (4096x1024)
  __bf16* xr_bf   = (__bf16*)(ws + 64 * MiB);   //  4 MiB (4096x512)
  __bf16* qw_bf   = (__bf16*)(ws + 68 * MiB);   // 0.5 MiB
  __bf16* kvw_bf  = (__bf16*)(ws + 68 * MiB + 512 * 1024);        // 1 MiB
  __bf16* pw_bf   = (__bf16*)(ws + 69 * MiB + 512 * 1024);        // 0.5 MiB
  __bf16* srw_bf  = (__bf16*)(ws + 70 * MiB);   // 2 MiB (2048x512)
  __bf16* net_bf  = patch;                      // reuse after conv GEMM

  // 1) convert inputs / weights to bf16
  f32_to_bf16<<<dim3(8192), dim3(256), 0, stream>>>(x, x_bf, BS * NQ * DIM);
  f32_to_bf16<<<dim3(256),  dim3(256), 0, stream>>>(q_w, qw_bf, DIM * DIM);
  f32_to_bf16<<<dim3(512),  dim3(256), 0, stream>>>(kv_w, kvw_bf, DIM * 2 * DIM);
  f32_to_bf16<<<dim3(256),  dim3(256), 0, stream>>>(proj_w, pw_bf, DIM * DIM);
  permute_srw<<<dim3(4096), dim3(256), 0, stream>>>(sr_w, srw_bf);

  // 2) q = (x @ q_w) * ATT_SCALE   (M=16384, K=512, N=512) -> bf16
  gemm_bf16<__bf16><<<dim3(8, 128), dim3(256), 0, stream>>>(
      x_bf, qw_bf, q_bf, nullptr, ATT_SCALE, BS * NQ, DIM, DIM);

  // 3) conv as patch-GEMM          (M=4096, K=2048, N=512) -> f32
  gather_patches<<<dim3(4096, 4), dim3(128), 0, stream>>>(x_bf, patch);
  gemm_bf16<float><<<dim3(8, 32), dim3(256), 0, stream>>>(
      patch, srw_bf, convout, nullptr, 1.0f, BS * NKV, DIM, 2048);

  // 4) LayerNorm(conv + sr_b) -> bf16
  layernorm_bias<<<dim3(4096), dim3(256), 0, stream>>>(convout, sr_b, ln_w,
                                                       ln_b, xr_bf);

  // 5) kv = xr @ kv_w              (M=4096, K=512, N=1024) -> bf16
  gemm_bf16<__bf16><<<dim3(16, 32), dim3(256), 0, stream>>>(
      xr_bf, kvw_bf, kv_bf, nullptr, 1.0f, BS * NKV, 2 * DIM, DIM);

  // 6) attention -> net_bf (reuses patch buffer)
  attention<<<dim3(NQ / 16, HEADS, BS), dim3(32), 0, stream>>>(q_bf, kv_bf,
                                                               net_bf);

  // 7) out = net @ proj_w + proj_b (M=16384, K=512, N=512) -> f32 d_out
  gemm_bf16<float><<<dim3(8, 128), dim3(256), 0, stream>>>(
      net_bf, pw_bf, out, proj_b, 1.0f, BS * NQ, DIM, DIM);
}
